// AdaptiveHierarchicalPooling_26766236188925
// MI455X (gfx1250) — matrix operand order repaired and checked by reading, not compiled, not run on hardware
//
#include <hip/hip_runtime.h>
#include <hip/hip_bf16.h>

// ---------------- problem constants (from reference) ----------------
#define BB   64            // graphs
#define NPG  4096          // nodes per graph (sorted equal segments)
#define HH   256           // hidden dim
#define CC0  256           // level-0 clusters
#define CC1  64            // level-1 clusters
#define NTOT (BB * NPG)    // 262144 nodes
#define KF   (CC1 * HH)    // 16384 = flattened f1 width

typedef __bf16 bf16;
typedef __attribute__((ext_vector_type(16))) __bf16 v16bf;
typedef __attribute__((ext_vector_type(8)))  __bf16 v8bf;
typedef __attribute__((ext_vector_type(8)))  float  v8f;
typedef __attribute__((ext_vector_type(4)))  float  v4f;

static __device__ __forceinline__ v8f wmma_bf16(v16bf a, v16bf b, v8f c) {
  // D = A(16x32) * B(32x16) + C, f32 accumulate
  return __builtin_amdgcn_wmma_f32_16x16x32_bf16(false, a, false, b, (short)0, c,
                                                 false, false);
}

// ---------------- WMMA fragment helpers (wave32) ----------------
// A 16x32 bf16 layout: lane L (row = L&15): halves = K[kb8..kb8+7], K[16+kb8..16+kb8+7]
// where kb8 = (L>>4)*8.  row_base must already include +kb8.
static __device__ __forceinline__ v16bf frag_a_bf16(const bf16* row_base) {
  v8bf lo = *(const v8bf*)row_base;
  v8bf hi = *(const v8bf*)(row_base + 16);
  v16bf a;
#pragma unroll
  for (int i = 0; i < 8; ++i) { a[i] = lo[i]; a[8 + i] = hi[i]; }
  return a;
}
static __device__ __forceinline__ v16bf frag_a_f32(const float* row_base) {
  v4f p0 = *(const v4f*)(row_base);
  v4f p1 = *(const v4f*)(row_base + 4);
  v4f p2 = *(const v4f*)(row_base + 16);
  v4f p3 = *(const v4f*)(row_base + 20);
  v16bf a;
#pragma unroll
  for (int i = 0; i < 4; ++i) {
    a[i]      = (bf16)p0[i];
    a[4 + i]  = (bf16)p1[i];
    a[8 + i]  = (bf16)p2[i];
    a[12 + i] = (bf16)p3[i];
  }
  return a;
}
// B 32x16 bf16 layout: lane L (col n = L&15): halves = K[kb8..kb8+7], K[16+kb8..16+kb8+7],
// kb8 = (L>>4)*8.  p must already include +kb8 and point into Bt row-major [n][k].
static __device__ __forceinline__ v16bf frag_b_bf16(const bf16* p) {
  v8bf lo = *(const v8bf*)p;
  v8bf hi = *(const v8bf*)(p + 16);
  v16bf b;
#pragma unroll
  for (int i = 0; i < 8; ++i) { b[i] = lo[i]; b[8 + i] = hi[i]; }
  return b;
}

// ---------------- kernel 0: weight transpose/convert ----------------
// W0t[c][k]=W0[k][c], W1t[c][k]=W1[k][c], Wft[h][k]=Wf[k][h]  (all bf16)
__global__ __launch_bounds__(256)
void k_prep(const float* __restrict__ W0, const float* __restrict__ W1,
            const float* __restrict__ Wf,
            bf16* __restrict__ W0t, bf16* __restrict__ W1t, bf16* __restrict__ Wft) {
  const int i = blockIdx.x * blockDim.x + threadIdx.x;
  if (i < CC0 * HH) {
    const int c = i / HH, k = i % HH;
    W0t[i] = (bf16)W0[(size_t)k * CC0 + c];
    return;
  }
  const int j = i - CC0 * HH;
  if (j < CC1 * HH) {
    const int c = j / HH, k = j % HH;
    W1t[j] = (bf16)W1[(size_t)k * CC1 + c];
    return;
  }
  const long l = (long)i - CC0 * HH - CC1 * HH;
  if (l >= 0 && l < (long)HH * KF) {
    const int h = (int)(l / KF), k = (int)(l % KF);
    Wft[l] = (bf16)Wf[(size_t)k * HH + h];
  }
}

// ---------------- kernel 1: A0 = softmax(x@W0 + b0), stored transposed ----
// One block = 32 nodes x all 256 clusters. 512 thr = 16 waves in a 2(M)x8(N) grid.
__global__ __launch_bounds__(512)
void k_assign0(const float* __restrict__ x, const bf16* __restrict__ W0t,
               const float* __restrict__ b0, bf16* __restrict__ A0t) {
  __shared__ float tile[32 * 257];   // padded stride to dodge bank conflicts
  __shared__ float rscale[32];
  const int tid  = threadIdx.x;
  const int lane = tid & 31;
  const int w    = tid >> 5;        // 0..15
  const int mi   = w & 1;           // node half
  const int nj   = w >> 1;          // 0..7 -> 32-col group
  const int node0 = blockIdx.x * 32;
  const int r    = lane & 15;
  const int kb8  = (lane >> 4) * 8;
  const int kb16 = (lane >> 4) * 16;

  v8f acc0 = {}, acc1 = {};
  const float* arow  = x + (size_t)(node0 + mi * 16 + r) * HH + kb8;
  const bf16*  brow0 = W0t + (size_t)(nj * 32 + r) * HH + kb8;
  const bf16*  brow1 = brow0 + 16 * HH;
#pragma unroll 4
  for (int k = 0; k < HH; k += 32) {
    v16bf a  = frag_a_f32(arow + k);
    acc0 = wmma_bf16(a, frag_b_bf16(brow0 + k), acc0);
    acc1 = wmma_bf16(a, frag_b_bf16(brow1 + k), acc1);
  }
  const float bias0 = b0[nj * 32 + r];
  const float bias1 = b0[nj * 32 + 16 + r];
  const int m0 = (lane >> 4) * 8;
#pragma unroll
  for (int v = 0; v < 8; ++v) {
    tile[(mi * 16 + m0 + v) * 257 + nj * 32 + r]      = acc0[v] + bias0;
    tile[(mi * 16 + m0 + v) * 257 + nj * 32 + 16 + r] = acc1[v] + bias1;
  }
  __syncthreads();

  { // row softmax over 256 cols: 16 threads per row + shfl-xor tree
    const int row = tid >> 4;
    const int j   = tid & 15;
    float mx = -1e30f;
    for (int c = j; c < CC0; c += 16) mx = fmaxf(mx, tile[row * 257 + c]);
#pragma unroll
    for (int m = 1; m < 16; m <<= 1) mx = fmaxf(mx, __shfl_xor(mx, m, 32));
    float s = 0.f;
    for (int c = j; c < CC0; c += 16) {
      float e = __expf(tile[row * 257 + c] - mx);
      tile[row * 257 + c] = e;
      s += e;
    }
#pragma unroll
    for (int m = 1; m < 16; m <<= 1) s += __shfl_xor(s, m, 32);
    if (j == 0) rscale[row] = 1.f / s;
  }
  __syncthreads();

  { // write A0t[g][c][n] bf16 (n contiguous -> coalesced 32B runs)
    const int g    = node0 / NPG;
    const int nloc = node0 % NPG;
    const int c    = tid >> 1;           // 0..255
    const int nb   = (tid & 1) * 16;
    bf16* op = A0t + ((size_t)(g * CC0 + c)) * NPG + nloc + nb;
#pragma unroll
    for (int i = 0; i < 16; ++i) {
      const int n = nb + i;
      op[i] = (bf16)(tile[n * 257 + c] * rscale[n]);
    }
  }
}

// ---------------- generic [z][rows][cols] f32 -> [z][cols][rows] bf16 -------
__global__ __launch_bounds__(256)
void k_transpose_f32_to_bf16(const float* __restrict__ in, bf16* __restrict__ out,
                             int rows, int cols) {
  __shared__ float t[32][33];
  const int r0 = blockIdx.x * 32;
  const int c0 = blockIdx.y * 32;
  const size_t base = (size_t)blockIdx.z * rows * cols;
  const int tid = threadIdx.x;
  const int lr = tid >> 3;            // 0..31
  const int lc = (tid & 7) * 4;
  v4f v = *(const v4f*)(in + base + (size_t)(r0 + lr) * cols + c0 + lc);
#pragma unroll
  for (int i = 0; i < 4; ++i) t[lr][lc + i] = v[i];
  __syncthreads();
  bf16* op = out + base + (size_t)(c0 + lr) * rows + r0 + lc;
#pragma unroll
  for (int i = 0; i < 4; ++i) op[i] = (bf16)t[lc + i][lr];
}

// ---------------- kernel 2: f0[g] = A0[g]^T @ x[g]  (256x256x4096 per graph)
// Block tile 64(M=c) x 128(N=h); 8 waves as 2x4; wave tile 32x32 (4 accums).
__global__ __launch_bounds__(256)
void k_pool0(const bf16* __restrict__ A0t, const bf16* __restrict__ xt,
             float* __restrict__ f0) {
  const int g  = blockIdx.z;
  const int cb = blockIdx.x * 64;
  const int hb = blockIdx.y * 128;
  const int tid = threadIdx.x, lane = tid & 31, w = tid >> 5;
  const int mi = w & 1, nj = w >> 1;                  // 2 x 4 waves
  const int r = lane & 15, kb8 = (lane >> 4) * 8, kb16 = (lane >> 4) * 16;
  const bf16* Ar0 = A0t + (size_t)(g * CC0 + cb + mi * 32 + r) * NPG + kb8;
  const bf16* Ar1 = Ar0 + (size_t)16 * NPG;
  const bf16* Br0 = xt + (size_t)(g * HH + hb + nj * 32 + r) * NPG + kb8;
  const bf16* Br1 = Br0 + (size_t)16 * NPG;
  v8f acc[2][2] = {};
  for (int k = 0; k < NPG; k += 32) {
    v16bf a0 = frag_a_bf16(Ar0 + k);
    v16bf a1 = frag_a_bf16(Ar1 + k);
    v16bf fb0 = frag_b_bf16(Br0 + k);
    v16bf fb1 = frag_b_bf16(Br1 + k);
    acc[0][0] = wmma_bf16(a0, fb0, acc[0][0]);
    acc[0][1] = wmma_bf16(a0, fb1, acc[0][1]);
    acc[1][0] = wmma_bf16(a1, fb0, acc[1][0]);
    acc[1][1] = wmma_bf16(a1, fb1, acc[1][1]);
  }
  const int n = lane & 15, m0 = (lane >> 4) * 8;
#pragma unroll
  for (int i = 0; i < 2; ++i)
#pragma unroll
    for (int j = 0; j < 2; ++j) {
      float* C = f0 + (size_t)(g * CC0 + cb + mi * 32 + i * 16) * HH
                    + hb + nj * 32 + j * 16;
#pragma unroll
      for (int v = 0; v < 8; ++v)
        C[(size_t)(m0 + v) * HH + n] = acc[i][j][v];
    }
}

// ---------------- kernel 3: A1 = softmax(f0@W1 + b1), stored transposed -----
// Block = 64 rows (c0 within one graph) x 64 cols; 8 waves as 4(M)x2(N).
__global__ __launch_bounds__(256)
void k_assign1(const float* __restrict__ f0, const bf16* __restrict__ W1t,
               const float* __restrict__ b1, bf16* __restrict__ A1t) {
  __shared__ float tile[64 * 65];
  __shared__ float rscale[64];
  const int tid = threadIdx.x, lane = tid & 31, w = tid >> 5;
  const int mi = w >> 1;               // 0..3 (16-row groups)
  const int nj = w & 1;                // 0..1 (32-col groups)
  const int row0 = blockIdx.x * 64;    // global f0 row (g*C0 + c0)
  const int r = lane & 15, kb8 = (lane >> 4) * 8, kb16 = (lane >> 4) * 16;
  const float* arow  = f0 + (size_t)(row0 + mi * 16 + r) * HH + kb8;
  const bf16*  brow0 = W1t + (size_t)(nj * 32 + r) * HH + kb8;
  const bf16*  brow1 = brow0 + 16 * HH;
  v8f acc0 = {}, acc1 = {};
#pragma unroll
  for (int k = 0; k < HH; k += 32) {
    v16bf a = frag_a_f32(arow + k);
    acc0 = wmma_bf16(a, frag_b_bf16(brow0 + k), acc0);
    acc1 = wmma_bf16(a, frag_b_bf16(brow1 + k), acc1);
  }
  const float bias0 = b1[nj * 32 + r];
  const float bias1 = b1[nj * 32 + 16 + r];
  const int m0 = (lane >> 4) * 8;
#pragma unroll
  for (int v = 0; v < 8; ++v) {
    tile[(mi * 16 + m0 + v) * 65 + nj * 32 + r]      = acc0[v] + bias0;
    tile[(mi * 16 + m0 + v) * 65 + nj * 32 + 16 + r] = acc1[v] + bias1;
  }
  __syncthreads();
  { // softmax over 64 cols: 4 threads per row
    const int row = tid >> 2;
    const int j   = tid & 3;
    float mx = -1e30f;
    for (int c = j; c < CC1; c += 4) mx = fmaxf(mx, tile[row * 65 + c]);
#pragma unroll
    for (int m = 1; m < 4; m <<= 1) mx = fmaxf(mx, __shfl_xor(mx, m, 32));
    float s = 0.f;
    for (int c = j; c < CC1; c += 4) {
      float e = __expf(tile[row * 65 + c] - mx);
      tile[row * 65 + c] = e;
      s += e;
    }
#pragma unroll
    for (int m = 1; m < 4; m <<= 1) s += __shfl_xor(s, m, 32);
    if (j == 0) rscale[row] = 1.f / s;
  }
  __syncthreads();
  { // A1t[g][c1][c0] bf16
    const int g   = row0 / CC0;
    const int c0b = row0 % CC0;
    const int c1  = tid >> 2;          // 0..63
    const int nb  = (tid & 3) * 16;
    bf16* op = A1t + (size_t)(g * CC1 + c1) * CC0 + c0b + nb;
#pragma unroll
    for (int i = 0; i < 16; ++i) {
      const int n = nb + i;
      op[i] = (bf16)(tile[n * 65 + c1] * rscale[n]);
    }
  }
}

// ---------------- kernel 4: f1[g] = A1[g]^T @ f0[g]  (64x256x256 per graph) -
__global__ __launch_bounds__(256)
void k_pool1(const bf16* __restrict__ A1t, const bf16* __restrict__ f0t,
             bf16* __restrict__ f1) {
  const int g  = blockIdx.z;
  const int hb = blockIdx.x * 128;
  const int tid = threadIdx.x, lane = tid & 31, w = tid >> 5;
  const int mi = w & 1, nj = w >> 1;
  const int r = lane & 15, kb8 = (lane >> 4) * 8, kb16 = (lane >> 4) * 16;
  const bf16* Ar0 = A1t + (size_t)(g * CC1 + mi * 32 + r) * CC0 + kb8;
  const bf16* Ar1 = Ar0 + (size_t)16 * CC0;
  const bf16* Br0 = f0t + (size_t)(g * HH + hb + nj * 32 + r) * CC0 + kb8;
  const bf16* Br1 = Br0 + (size_t)16 * CC0;
  v8f acc[2][2] = {};
#pragma unroll
  for (int k = 0; k < CC0; k += 32) {
    v16bf a0 = frag_a_bf16(Ar0 + k);
    v16bf a1 = frag_a_bf16(Ar1 + k);
    v16bf fb0 = frag_b_bf16(Br0 + k);
    v16bf fb1 = frag_b_bf16(Br1 + k);
    acc[0][0] = wmma_bf16(a0, fb0, acc[0][0]);
    acc[0][1] = wmma_bf16(a0, fb1, acc[0][1]);
    acc[1][0] = wmma_bf16(a1, fb0, acc[1][0]);
    acc[1][1] = wmma_bf16(a1, fb1, acc[1][1]);
  }
  const int n = lane & 15, m0 = (lane >> 4) * 8;
#pragma unroll
  for (int i = 0; i < 2; ++i)
#pragma unroll
    for (int j = 0; j < 2; ++j) {
      bf16* C = f1 + (size_t)g * KF + (size_t)(mi * 32 + i * 16) * HH
                   + hb + nj * 32 + j * 16;
#pragma unroll
      for (int v = 0; v < 8; ++v)
        C[(size_t)(m0 + v) * HH + n] = (bf16)acc[i][j][v];
    }
}

// ---------------- kernel 5: out = relu(f1 @ Wf + bf)  (64x256x16384) --------
__global__ __launch_bounds__(256)
void k_final(const bf16* __restrict__ f1, const bf16* __restrict__ Wft,
             const float* __restrict__ bfb, float* __restrict__ out) {
  const int tid = threadIdx.x, lane = tid & 31, w = tid >> 5;
  const int mt = w & 3;                 // 4 x 16 = 64 graph rows
  const int nt = w >> 2;                // 0..1
  const int hb = blockIdx.x * 32 + nt * 16;
  const int r = lane & 15, kb8 = (lane >> 4) * 8, kb16 = (lane >> 4) * 16;
  const bf16* Ab = f1 + (size_t)(mt * 16 + r) * KF + kb8;
  const bf16* Bb = Wft + (size_t)(hb + r) * KF + kb8;
  v8f acc = {};
#pragma unroll 4
  for (int k = 0; k < KF; k += 32)
    acc = wmma_bf16(frag_a_bf16(Ab + k), frag_b_bf16(Bb + k), acc);
  const int n = lane & 15, m0 = (lane >> 4) * 8;
  const float bias = bfb[hb + n];
#pragma unroll
  for (int v = 0; v < 8; ++v) {
    float val = acc[v] + bias;
    out[(size_t)(mt * 16 + m0 + v) * HH + hb + n] = val > 0.f ? val : 0.f;
  }
}

// ---------------- launch ----------------------------------------------------
extern "C" void kernel_launch(void* const* d_in, const int* in_sizes, int n_in,
                              void* d_out, int out_size, void* d_ws, size_t ws_size,
                              hipStream_t stream) {
  const float* x   = (const float*)d_in[0];
  // d_in[1] edge_index (int64) and d_in[2] batch (int64) are unused: batch ids
  // are sorted, equal-size segments per the reference.
  const float* W0  = (const float*)d_in[3];
  const float* b0  = (const float*)d_in[4];
  const float* W1  = (const float*)d_in[5];
  const float* b1  = (const float*)d_in[6];
  const float* Wf  = (const float*)d_in[7];
  const float* bfb = (const float*)d_in[8];
  float* out = (float*)d_out;
  (void)in_sizes; (void)n_in; (void)out_size; (void)ws_size;

  // workspace carve-up (~306 MB total)
  char* ws = (char*)d_ws;
  size_t off = 0;
  auto take = [&](size_t bytes) {
    char* p = ws + off;
    off = (off + bytes + 255) & ~(size_t)255;
    return p;
  };
  bf16*  W0t = (bf16*)take((size_t)CC0 * HH * 2);        // 128 KB
  bf16*  W1t = (bf16*)take((size_t)CC1 * HH * 2);        // 32 KB
  bf16*  Wft = (bf16*)take((size_t)HH * KF * 2);         // 8 MB
  bf16*  A0t = (bf16*)take((size_t)BB * CC0 * NPG * 2);  // 128 MB
  bf16*  xt  = (bf16*)take((size_t)BB * HH * NPG * 2);   // 128 MB
  float* f0  = (float*)take((size_t)BB * CC0 * HH * 4);  // 16 MB
  bf16*  f0t = (bf16*)take((size_t)BB * HH * CC0 * 2);   // 8 MB
  bf16*  A1t = (bf16*)take((size_t)BB * CC1 * CC0 * 2);  // 2 MB
  bf16*  f1  = (bf16*)take((size_t)BB * KF * 2);         // 2 MB

  const int prep_total = CC0 * HH + CC1 * HH + HH * KF;
  k_prep<<<(prep_total + 255) / 256, 256, 0, stream>>>(W0, W1, Wf, W0t, W1t, Wft);
  k_assign0<<<NTOT / 32, 512, 0, stream>>>(x, W0t, b0, A0t);
  k_transpose_f32_to_bf16<<<dim3(NPG / 32, HH / 32, BB), 256, 0, stream>>>(x, xt, NPG, HH);
  k_pool0<<<dim3(CC0 / 64, HH / 128, BB), 256, 0, stream>>>(A0t, xt, f0);
  k_transpose_f32_to_bf16<<<dim3(CC0 / 32, HH / 32, BB), 256, 0, stream>>>(f0, f0t, CC0, HH);
  k_assign1<<<(BB * CC0) / 64, 256, 0, stream>>>(f0, W1t, b1, A1t);
  k_pool1<<<dim3(HH / 128, 1, BB), 256, 0, stream>>>(A1t, f0t, f1);
  k_final<<<HH / 32, 256, 0, stream>>>(f1, Wft, bfb, out);
}
